// SparseFlashAttention_12120397709557
// MI455X (gfx1250) — matrix operand order repaired and checked by reading, not compiled, not run on hardware
//
#include <hip/hip_runtime.h>
#include <hip/hip_bf16.h>

// Sparse (masked) flash attention for B=2, S=2048, H=16, D=64, fp32 I/O.
// Pass 1: convert K,V to bf16 once (memory-bound, ~µs).
// Pass 2: flash attention with v_wmma_f32_16x16x32_bf16, online softmax.
//         K chunk staged into LDS via TDM tensor_load_to_lds (TENSORcnt),
//         V chunk staged transposed by the block, P transposed via LDS.

#define B_  2
#define S_  2048
#define H_  16
#define D_  64
#define WPB 4   // waves per block (each wave owns one 16-query tile)
#define NELEM (B_ * S_ * H_ * D_)   // 4194304 per tensor

typedef __attribute__((ext_vector_type(16))) __bf16 v16bf;
typedef __attribute__((ext_vector_type(8)))  float  v8f;
typedef __attribute__((ext_vector_type(4)))  unsigned int v4u;
typedef __attribute__((ext_vector_type(8)))  int    v8i;
typedef __attribute__((ext_vector_type(4)))  int    v4i;

union Frag {
    v16bf          v;
    __bf16         h[16];
    unsigned short s[16];
    uint4          q[2];
};

__device__ __forceinline__ __bf16 f2bf(float f) { return (__bf16)f; }
__device__ __forceinline__ unsigned short f2bfu(float f) {
    __bf16 h = (__bf16)f;
    return __builtin_bit_cast(unsigned short, h);
}

__device__ __forceinline__ float red_max16(float v) {
    v = fmaxf(v, __shfl_xor(v, 1, 32));
    v = fmaxf(v, __shfl_xor(v, 2, 32));
    v = fmaxf(v, __shfl_xor(v, 4, 32));
    v = fmaxf(v, __shfl_xor(v, 8, 32));
    return v;
}
__device__ __forceinline__ float red_sum16(float v) {
    v += __shfl_xor(v, 1, 32);
    v += __shfl_xor(v, 2, 32);
    v += __shfl_xor(v, 4, 32);
    v += __shfl_xor(v, 8, 32);
    return v;
}

// ---- Pass 1: f32 -> bf16, 8 elements per thread ----
__global__ __launch_bounds__(256, 4)
void cvt_f32_bf16(const float* __restrict__ src, unsigned short* __restrict__ dst) {
    const int i = blockIdx.x * blockDim.x + threadIdx.x;   // 0 .. NELEM/8-1
    const float4* s = (const float4*)src + (size_t)i * 2;
    float4 a = s[0], b = s[1];
    union { uint4 q; unsigned short u[8]; } t;
    t.u[0] = f2bfu(a.x); t.u[1] = f2bfu(a.y); t.u[2] = f2bfu(a.z); t.u[3] = f2bfu(a.w);
    t.u[4] = f2bfu(b.x); t.u[5] = f2bfu(b.y); t.u[6] = f2bfu(b.z); t.u[7] = f2bfu(b.w);
    ((uint4*)dst)[i] = t.q;
}

// ---- Pass 2: flash attention ----
__global__ __launch_bounds__(WPB * 32, 1)
void sparse_flash_attn_wmma(const float* __restrict__ q,
                            const unsigned short* __restrict__ k16,
                            const unsigned short* __restrict__ v16,
                            const unsigned char* __restrict__ mask,
                            float* __restrict__ out) {
    __shared__ __align__(16) unsigned short lds_k [32][64];      // [key][d]   4 KB (TDM dest)
    __shared__ __align__(16) unsigned short lds_vt[64][32];      // [d][key]   4 KB
    __shared__ __align__(16) unsigned short lds_p [WPB][16][32]; // P tiles    4 KB

    const int tid  = threadIdx.x;
    const int lane = tid & 31;
    const int wave = tid >> 5;
    const int half = lane >> 4;     // 0: lanes 0-15, 1: lanes 16-31
    const int ln   = lane & 15;

    const int h     = blockIdx.y;
    const int b     = blockIdx.z;
    const int qbase = (blockIdx.x * WPB + wave) * 16;

    const float scale = 0.125f;  // 1/sqrt(64)

    // Staging coordinates: each thread owns (key row, 16-wide d segment).
    const int srow = tid >> 2;        // 0..31 key within chunk
    const int sseg = (tid & 3) * 16;  // d segment base: 0,16,32,48

    const unsigned lds_k_off = (unsigned)(uintptr_t)&lds_k[0][0];  // LDS byte offset

    // ---- Q A-fragments (one per 32-wide d-chunk), converted to bf16 ----
    const float* qrow = q + (((size_t)b * S_ + (qbase + ln)) * H_ + h) * D_;
    Frag aq[2];
#pragma unroll
    for (int dc = 0; dc < 2; ++dc) {
#pragma unroll
        for (int j = 0; j < 8; ++j) {
            aq[dc].h[j]     = f2bf(qrow[dc * 32 + half * 8 + j]);
            aq[dc].h[8 + j] = f2bf(qrow[dc * 32 + 16 + half * 8 + j]);
        }
    }

    v8f o[4];
#pragma unroll
    for (int dt = 0; dt < 4; ++dt)
#pragma unroll
        for (int r = 0; r < 8; ++r) o[dt][r] = 0.0f;

    float run_max[8], run_sum[8];
#pragma unroll
    for (int r = 0; r < 8; ++r) { run_max[r] = -INFINITY; run_sum[r] = 0.0f; }

    for (int kb = 0; kb < S_; kb += 32) {
        if (kb + 32 < S_) {
            const size_t nrow = (((size_t)b * S_ + (kb + 32 + srow)) * H_ + h) * D_ + sseg;
            __builtin_prefetch(k16 + nrow, 0, 0);
            __builtin_prefetch(v16 + nrow, 0, 0);
        }

        __syncthreads();  // previous iteration's readers done before restaging

        // ---- TDM: async 2D tile copy of K chunk (32 keys x 64 d, bf16) ----
        if (wave == 0) {
            const unsigned long long gaddr =
                (unsigned long long)(uintptr_t)(k16 + (((size_t)b * S_ + kb) * H_ + h) * D_);
            unsigned g0x = 1u;  // count=1 (valid descriptor)
            unsigned g0y = (unsigned)__builtin_amdgcn_readfirstlane((int)lds_k_off);
            unsigned g0z = (unsigned)__builtin_amdgcn_readfirstlane((int)(gaddr & 0xFFFFFFFFull));
            unsigned g0w = ((unsigned)__builtin_amdgcn_readfirstlane(
                               (int)((gaddr >> 32) & 0x01FFFFFFull)))
                           | (2u << 30);  // type=2 ("image")
            v4u g0 = { g0x, g0y, g0z, g0w };
            // group1: data_size=2B (code 1); tensor_dim0=64; tensor_dim1=32;
            // tile_dim0=64; tile_dim1=32; tensor_dim0_stride=H*D=1024.
            v8i g1 = { (int)0x00010000,          // wg_mask=0, data_size=1
                       (int)(64u << 16),         // tensor_dim0 low16 @ bits63:48
                       (int)(32u << 16),         // tensor_dim1 low16 @ bits95:80
                       (int)(64u << 16),         // tile_dim0 @ bits127:112
                       (int)32,                  // tile_dim1 @ bits143:128
                       (int)(H_ * D_),           // tensor_dim0_stride low32
                       0, 0 };
            v4i gz = { 0, 0, 0, 0 };
            v8i gz8 = { 0, 0, 0, 0, 0, 0, 0, 0 };
            __builtin_amdgcn_tensor_load_to_lds(g0, g1, gz, gz, gz8, 0);
        }

        // ---- Cooperative stage: V chunk transposed (bf16 -> bf16, no cvt) ----
        {
            const unsigned short* vrow =
                v16 + (((size_t)b * S_ + (kb + srow)) * H_ + h) * D_ + sseg;
            union { uint4 q[2]; unsigned short u[16]; } t;
            t.q[0] = *(const uint4*)vrow;
            t.q[1] = *(const uint4*)(vrow + 8);
#pragma unroll
            for (int j = 0; j < 16; ++j)
                lds_vt[sseg + j][srow] = t.u[j];
        }

        if (wave == 0) __builtin_amdgcn_s_wait_tensorcnt(0);
        __syncthreads();  // staged K (TDM) + V visible to all 4 waves

        // ---- Scores: S = Q * K^T (2 key tiles of 16) ----
        float st[2][8];
#pragma unroll
        for (int t = 0; t < 2; ++t) {
            v8f sc = {0.f, 0.f, 0.f, 0.f, 0.f, 0.f, 0.f, 0.f};
#pragma unroll
            for (int dc = 0; dc < 2; ++dc) {
                Frag bk;
                bk.q[0] = *(const uint4*)&lds_k[t * 16 + ln][dc * 32 + half * 8];
                bk.q[1] = *(const uint4*)&lds_k[t * 16 + ln][dc * 32 + 16 + half * 8];
                sc = __builtin_amdgcn_wmma_f32_16x16x32_bf16(
                    false, aq[dc].v, false, bk.v, (short)0, sc, false, false);
            }
#pragma unroll
            for (int r = 0; r < 8; ++r) {
                const size_t mrow = (size_t)(qbase + half * 8 + r) * S_;
                unsigned char mb  = mask[mrow + kb + t * 16 + ln];
                st[t][r] = mb ? sc[r] * scale : -1.0e30f;
            }
        }

        // ---- Online softmax over this 32-key chunk ----
        float alpha[8];
#pragma unroll
        for (int r = 0; r < 8; ++r) {
            float cm = red_max16(fmaxf(st[0][r], st[1][r]));
            float nm = fmaxf(run_max[r], cm);
            alpha[r] = __expf(run_max[r] - nm);
            run_max[r] = nm;
            float p0 = __expf(st[0][r] - nm);
            float p1 = __expf(st[1][r] - nm);
            st[0][r] = p0; st[1][r] = p1;
            run_sum[r] = run_sum[r] * alpha[r] + red_sum16(p0 + p1);
        }

#pragma unroll
        for (int dt = 0; dt < 4; ++dt)
#pragma unroll
            for (int r = 0; r < 8; ++r) o[dt][r] *= alpha[r];

        // ---- Transpose P from C layout to A layout via per-wave LDS ----
#pragma unroll
        for (int r = 0; r < 8; ++r) {
            lds_p[wave][half * 8 + r][ln]      = f2bfu(st[0][r]);
            lds_p[wave][half * 8 + r][16 + ln] = f2bfu(st[1][r]);
        }
        asm volatile("s_wait_dscnt 0" ::: "memory");

        Frag ap;
        ap.q[0] = *(const uint4*)&lds_p[wave][ln][half * 8];
        ap.q[1] = *(const uint4*)&lds_p[wave][ln][16 + half * 8];

        // ---- O += P * V ----
#pragma unroll
        for (int dt = 0; dt < 4; ++dt) {
            Frag bv;
            bv.q[0] = *(const uint4*)&lds_vt[dt * 16 + ln][half * 8];
            bv.q[1] = *(const uint4*)&lds_vt[dt * 16 + ln][16 + half * 8];
            o[dt] = __builtin_amdgcn_wmma_f32_16x16x32_bf16(
                false, ap.v, false, bv.v, (short)0, o[dt], false, false);
        }
    }

    // ---- Epilogue ----
    float inv[8];
#pragma unroll
    for (int r = 0; r < 8; ++r)
        inv[r] = (run_max[r] > -1.0e29f) ? (1.0f / run_sum[r]) : 0.0f;

#pragma unroll
    for (int dt = 0; dt < 4; ++dt) {
#pragma unroll
        for (int r = 0; r < 8; ++r) {
            const size_t off =
                (((size_t)b * S_ + (qbase + half * 8 + r)) * H_ + h) * D_ + dt * 16 + ln;
            out[off] = o[dt][r] * inv[r];
        }
    }
}

extern "C" void kernel_launch(void* const* d_in, const int* in_sizes, int n_in,
                              void* d_out, int out_size, void* d_ws, size_t ws_size,
                              hipStream_t stream) {
    (void)in_sizes; (void)n_in; (void)ws_size; (void)out_size;
    const float*         q    = (const float*)d_in[0];
    const float*         k    = (const float*)d_in[1];
    const float*         v    = (const float*)d_in[2];
    const unsigned char* mask = (const unsigned char*)d_in[3];
    float*               out  = (float*)d_out;

    unsigned short* k16 = (unsigned short*)d_ws;           // NELEM bf16 (8 MB)
    unsigned short* v16 = k16 + (size_t)NELEM;             // NELEM bf16 (8 MB)

    // Pass 1: one-time K/V conversion to bf16.
    cvt_f32_bf16<<<NELEM / 8 / 256, 256, 0, stream>>>(k, k16);
    cvt_f32_bf16<<<NELEM / 8 / 256, 256, 0, stream>>>(v, v16);

    // Pass 2: attention.
    dim3 grid(S_ / 16 / WPB, H_, B_);   // (32, 16, 2)
    dim3 block(WPB * 32);               // 4 waves of 32
    sparse_flash_attn_wmma<<<grid, block, 0, stream>>>(q, k16, v16, mask, out);
}
